// LineFeatureAggregator_51926154609362
// MI455X (gfx1250) — matrix operand order repaired and checked by reading, not compiled, not run on hardware
//
#include <hip/hip_runtime.h>
#include <hip/hip_bf16.h>

// Problem constants (match reference)
#define BB   8
#define SS   4096
#define HH   768
#define LL   256
#define AH   1024
#define NTOK (BB * SS)          // 32768 token rows
#define FEAT_OUT ((size_t)BB * LL * HH)

#define ALO  (16 * HH)          // lo-plane offset (elements) within LDS A buffer
#define BLO  (AH * HH)          // lo-plane offset (elements) within W1t buffer

typedef __attribute__((ext_vector_type(16))) __bf16 v16bf;
typedef __attribute__((ext_vector_type(8)))  float  v8f;
typedef __attribute__((ext_vector_type(4)))  unsigned int u32x4;

union FragBF { v16bf v; u32x4 q[2]; };

__device__ __forceinline__ unsigned short bf16_rne(float x) {
    unsigned int u = __float_as_uint(x);
    unsigned int r = u + 0x7FFFu + ((u >> 16) & 1u);
    return (unsigned short)(r >> 16);
}
__device__ __forceinline__ float bf16_to_f(unsigned short h) {
    return __uint_as_float(((unsigned int)h) << 16);
}

// ---------------------------------------------------------------------------
// Kernel 0: transpose W1 [H, AH] (fp32) -> W1t [AH, H] split into bf16 hi/lo
// (hi plane at offset 0, lo plane at offset BLO elements)
// ---------------------------------------------------------------------------
__global__ void conv_w1_kernel(const float* __restrict__ W1,
                               unsigned short* __restrict__ W1t) {
    int e = blockIdx.x * 256 + threadIdx.x;
    if (e >= AH * HH) return;
    int n = e / HH;
    int k = e % HH;
    float x = W1[(size_t)k * AH + n];
    unsigned short h = bf16_rne(x);
    W1t[e] = h;
    W1t[e + BLO] = bf16_rne(x - bf16_to_f(h));
}

// ---------------------------------------------------------------------------
// Kernel 1: fused scores = tanh(X @ W1 + b1) @ W2 + b2 via bf16 WMMA
// Grid: NTOK/16 blocks of 256 threads (8 waves). Block handles 16 token rows;
// wave w owns hidden columns [w*128, w*128+128) (8 N-tiles of 16).
// hi/lo bf16 split => 3 WMMAs per (k-step, tile), interleaved over tile pairs.
// All B loads use one per-lane pointer + compile-time immediate offsets.
// ---------------------------------------------------------------------------
__global__ __launch_bounds__(256) void scores_kernel(
        const float* __restrict__ X,
        const unsigned short* __restrict__ W1t,
        const float* __restrict__ b1,
        const float* __restrict__ W2,
        const float* __restrict__ b2,
        float* __restrict__ scores) {
    __shared__ unsigned short As[2 * 16 * HH];   // hi plane + lo plane, 48 KB
    __shared__ float part[8][16];

    const int tid = threadIdx.x;
    const int m0  = blockIdx.x * 16;

    // Stage 16x768 fp32 tile -> bf16 hi/lo planes in LDS
    const float* Xt = X + (size_t)m0 * HH;
    for (int i = tid; i < 16 * HH; i += 256) {
        float x = Xt[i];
        unsigned short h = bf16_rne(x);
        As[i] = h;
        As[i + ALO] = bf16_rne(x - bf16_to_f(h));
    }
    __syncthreads();

    const int lane = tid & 31;
    const int wave = tid >> 5;
    const int lh   = lane & 15;   // A: row in tile / B,C: n within tile
    const int half = lane >> 4;
    const int n0   = wave * 128;

    v8f acc[8];
#pragma unroll
    for (int t = 0; t < 8; ++t)
#pragma unroll
        for (int i = 0; i < 8; ++i) acc[t][i] = 0.0f;

    // Per-lane bases; everything else is an immediate offset.
    // A fragment: lanes 0-15 K=k0..k0+15, lanes 16-31 K=k0+16..k0+31
    // B fragment: lanes 0-15 K=k0..k0+15, lanes 16-31 K=k0+16..k0+31
    unsigned aoff = (unsigned)(lh * HH + half * 16);
    const unsigned short* bptr = W1t + (unsigned)(n0 + lh) * HH + half * 16;

    for (int k0 = 0; k0 < HH; k0 += 32) {
        FragBF ah, al;
        ah.q[0] = *(const u32x4*)&As[aoff];
        ah.q[1] = *(const u32x4*)&As[aoff + 8];
        al.q[0] = *(const u32x4*)&As[aoff + ALO];
        al.q[1] = *(const u32x4*)&As[aoff + ALO + 8];
#pragma unroll
        for (int t = 0; t < 8; t += 2) {
            const v16bf bh0 = *(const v16bf*)(bptr + (t    ) * 16 * HH);
            const v16bf bh1 = *(const v16bf*)(bptr + (t + 1) * 16 * HH);
            const v16bf bl0 = *(const v16bf*)(bptr + (t    ) * 16 * HH + BLO);
            const v16bf bl1 = *(const v16bf*)(bptr + (t + 1) * 16 * HH + BLO);
            acc[t]     = __builtin_amdgcn_wmma_f32_16x16x32_bf16(
                false, ah.v, false, bh0, (short)0, acc[t],     false, false);
            acc[t + 1] = __builtin_amdgcn_wmma_f32_16x16x32_bf16(
                false, ah.v, false, bh1, (short)0, acc[t + 1], false, false);
            acc[t]     = __builtin_amdgcn_wmma_f32_16x16x32_bf16(
                false, al.v, false, bh0, (short)0, acc[t],     false, false);
            acc[t + 1] = __builtin_amdgcn_wmma_f32_16x16x32_bf16(
                false, al.v, false, bh1, (short)0, acc[t + 1], false, false);
            acc[t]     = __builtin_amdgcn_wmma_f32_16x16x32_bf16(
                false, ah.v, false, bl0, (short)0, acc[t],     false, false);
            acc[t + 1] = __builtin_amdgcn_wmma_f32_16x16x32_bf16(
                false, ah.v, false, bl1, (short)0, acc[t + 1], false, false);
        }
        aoff += 32;
        bptr += 32;
    }

    // Epilogue: + b1, tanh, * W2, partial row sums.
    // C layout: VGPR i, lanes 0-15 -> row i, n=lane; lanes 16-31 -> row i+8
    float rowsum[8];
#pragma unroll
    for (int i = 0; i < 8; ++i) rowsum[i] = 0.0f;
#pragma unroll
    for (int t = 0; t < 8; ++t) {
        const int n = n0 + t * 16 + lh;
        const float bb = b1[n];
        const float ww = W2[n];
#pragma unroll
        for (int i = 0; i < 8; ++i)
            rowsum[i] += tanhf(acc[t][i] + bb) * ww;
    }
    // Reduce across the 16 lanes of each half (xor masks stay within half)
#pragma unroll
    for (int i = 0; i < 8; ++i) {
        rowsum[i] += __shfl_xor(rowsum[i], 1, 32);
        rowsum[i] += __shfl_xor(rowsum[i], 2, 32);
        rowsum[i] += __shfl_xor(rowsum[i], 4, 32);
        rowsum[i] += __shfl_xor(rowsum[i], 8, 32);
    }
    if (lh == 0) {
#pragma unroll
        for (int i = 0; i < 8; ++i) part[wave][half * 8 + i] = rowsum[i];
    }
    __syncthreads();
    if (tid < 16) {
        float s = b2[0];
#pragma unroll
        for (int w = 0; w < 8; ++w) s += part[w][tid];
        scores[m0 + tid] = s;
    }
}

// ---------------------------------------------------------------------------
// Kernel 2: per-(batch,line) segment softmax + weighted feature sum.
// Grid: B*L blocks of 256 threads. Fully deterministic (prefix-scan
// compaction, fixed-order reductions, no atomics).
// ---------------------------------------------------------------------------
__global__ __launch_bounds__(256) void agg_kernel(
        const float* __restrict__ X,
        const int* __restrict__ line_ids,
        const float* __restrict__ scores,
        float* __restrict__ out) {
    __shared__ int   idx[SS];     // 16 KB
    __shared__ float ev[SS];      // 16 KB
    __shared__ int   cnt[256];
    __shared__ int   offs[257];
    __shared__ float red[256];
    __shared__ float bc[2];

    const int b = blockIdx.x / LL;
    const int l = blockIdx.x % LL;
    const int t = threadIdx.x;
    const int* ids = line_ids + (size_t)b * SS;

    // Count matches in this thread's contiguous 16-token chunk
    int c = 0;
#pragma unroll
    for (int j = 0; j < 16; ++j) c += (ids[t * 16 + j] == l) ? 1 : 0;
    cnt[t] = c;
    __syncthreads();
    if (t == 0) {
        int s = 0;
        for (int i = 0; i < 256; ++i) { offs[i] = s; s += cnt[i]; }
        offs[256] = s;
    }
    __syncthreads();
    const int total = offs[256];
    const size_t fbase = (size_t)blockIdx.x * HH;

    if (total == 0) {
        out[fbase + t]       = 0.0f;
        out[fbase + t + 256] = 0.0f;
        out[fbase + t + 512] = 0.0f;
        if (t == 0) out[FEAT_OUT + blockIdx.x] = 0.0f;  // mask = False
        return;
    }

    // Ordered compaction of matching token indices
    {
        int k = offs[t];
        for (int j = 0; j < 16; ++j) {
            int tok = t * 16 + j;
            if (ids[tok] == l) idx[k++] = tok;
        }
    }
    __syncthreads();

    const float* sc = scores + (size_t)b * SS;
    // Segment max
    float m = -INFINITY;
    for (int p = t; p < total; p += 256) m = fmaxf(m, sc[idx[p]]);
    red[t] = m;
    __syncthreads();
    for (int st = 128; st > 0; st >>= 1) {
        if (t < st) red[t] = fmaxf(red[t], red[t + st]);
        __syncthreads();
    }
    if (t == 0) bc[0] = red[0];
    __syncthreads();
    const float maxv = bc[0];

    // exp + denom
    float s = 0.0f;
    for (int p = t; p < total; p += 256) {
        float e = expf(sc[idx[p]] - maxv);
        ev[p] = e;
        s += e;
    }
    red[t] = s;
    __syncthreads();
    for (int st = 128; st > 0; st >>= 1) {
        if (t < st) red[t] += red[t + st];
        __syncthreads();
    }
    if (t == 0) bc[1] = 1.0f / fmaxf(red[0], 1e-20f);
    __syncthreads();
    const float inv = bc[1];

    // Weighted sum of token rows; thread t owns feature dims t, t+256, t+512
    float a0 = 0.0f, a1 = 0.0f, a2 = 0.0f;
    for (int p = 0; p < total; ++p) {
        const float w = ev[p] * inv;
        const float* row = X + ((size_t)b * SS + idx[p]) * HH;
        a0 += w * row[t];
        a1 += w * row[t + 256];
        a2 += w * row[t + 512];
    }
    out[fbase + t]       = a0;
    out[fbase + t + 256] = a1;
    out[fbase + t + 512] = a2;
    if (t == 0) out[FEAT_OUT + blockIdx.x] = 1.0f;  // mask = True
}

// ---------------------------------------------------------------------------
extern "C" void kernel_launch(void* const* d_in, const int* in_sizes, int n_in,
                              void* d_out, int out_size, void* d_ws, size_t ws_size,
                              hipStream_t stream) {
    (void)in_sizes; (void)n_in; (void)out_size; (void)ws_size;
    const float* token_hidden = (const float*)d_in[0];
    const int*   line_ids     = (const int*)d_in[1];
    const float* W1           = (const float*)d_in[2];
    const float* b1           = (const float*)d_in[3];
    const float* W2           = (const float*)d_in[4];
    const float* b2           = (const float*)d_in[5];
    float* out = (float*)d_out;

    // Workspace layout: W1t hi+lo planes (3 MB) | scores (128 KB)
    unsigned short* W1t = (unsigned short*)d_ws;
    float* scores = (float*)(W1t + (size_t)2 * AH * HH);

    conv_w1_kernel<<<(AH * HH + 255) / 256, 256, 0, stream>>>(W1, W1t);
    scores_kernel<<<NTOK / 16, 256, 0, stream>>>(token_hidden, W1t,
                                                 b1, W2, b2, scores);
    agg_kernel<<<BB * LL, 256, 0, stream>>>(token_hidden, line_ids, scores, out);
}
